// MultiHeadSelfAttention_27341761806622
// MI455X (gfx1250) — matrix operand order repaired and checked by reading, not compiled, not run on hardware
//
#include <hip/hip_runtime.h>

// ---------------------------------------------------------------------------
// MI455X / gfx1250 multi-head self-attention (B=8, S=1024, D=768, H=12, Hd=64)
// bf16 WMMA (v_wmma_f32_16x16x32_bf16) everywhere, f32 accumulate.
// Async global->LDS staging (double buffered) + DPP softmax reductions.
// sched_group_barrier pipelines DS fragment loads against WMMA issue.
// ---------------------------------------------------------------------------

typedef __bf16 bf16_t;
typedef __attribute__((ext_vector_type(4)))  __bf16 v4bf;
typedef __attribute__((ext_vector_type(8)))  __bf16 v8bf;
typedef __attribute__((ext_vector_type(16))) __bf16 v16bf;
typedef __attribute__((ext_vector_type(4)))  float  v4f;
typedef __attribute__((ext_vector_type(8)))  float  v8f;
typedef __attribute__((ext_vector_type(4)))  int    v4i_t;

#define NB   8
#define NS   1024
#define ND   768
#define NH   12
#define NHD  64
#define NM   (NB * NS)          // 8192 rows

#if __has_builtin(__builtin_amdgcn_global_load_async_to_lds_b128)
#define HAS_ASYNC_LDS 1
typedef __attribute__((address_space(1))) v4i_t* gas_v4i_p;   // global
typedef __attribute__((address_space(3))) v4i_t* las_v4i_p;   // LDS
#else
#define HAS_ASYNC_LDS 0
#endif

// DS-read / matrix-op pipeline hint: DSx2, (WMMA, DSx2)x3, WMMA
__device__ __forceinline__ void pipeline_ds_wmma() {
#if __has_builtin(__builtin_amdgcn_sched_group_barrier)
  __builtin_amdgcn_sched_group_barrier(0x100, 2, 0);   // 2 DS reads
#pragma unroll
  for (int i = 0; i < 3; ++i) {
    __builtin_amdgcn_sched_group_barrier(0x008, 1, 0); // 1 WMMA
    __builtin_amdgcn_sched_group_barrier(0x100, 2, 0); // 2 DS reads
  }
  __builtin_amdgcn_sched_group_barrier(0x008, 1, 0);   // last WMMA
#endif
}

__device__ __forceinline__ v16bf pack16(v8bf lo, v8bf hi) {
  return __builtin_shufflevector(lo, hi, 0,1,2,3,4,5,6,7,8,9,10,11,12,13,14,15);
}

__device__ __forceinline__ v16bf load_frag(const bf16_t* p) {
  return pack16(*reinterpret_cast<const v8bf*>(p),
                *reinterpret_cast<const v8bf*>(p + 16));
}

__device__ __forceinline__ v8f wmma_bf16(v16bf a, v16bf b, v8f c) {
  // D = A(16x32 bf16) x B(32x16 bf16) + C(16x16 f32)
  return __builtin_amdgcn_wmma_f32_16x16x32_bf16(false, a, false, b, (short)0, c,
                                                 false, false);
}

// 16-byte global -> LDS copy; async on CDNA5, synchronous fallback otherwise.
__device__ __forceinline__ void copy16_g2l(bf16_t* lds, const bf16_t* g) {
#if HAS_ASYNC_LDS
  __builtin_amdgcn_global_load_async_to_lds_b128(
      (gas_v4i_p)(void*)g, (las_v4i_p)(void*)lds, 0, 0);
#else
  *reinterpret_cast<v8bf*>(lds) = *reinterpret_cast<const v8bf*>(g);
#endif
}

__device__ __forceinline__ void wait_async_lds() {
#if HAS_ASYNC_LDS
#if __has_builtin(__builtin_amdgcn_s_wait_asynccnt)
  __builtin_amdgcn_s_wait_asynccnt(0);
#else
  asm volatile("s_wait_asynccnt 0" ::: "memory");
#endif
#endif
}

// ----- DPP butterfly reduction over each 16-lane half (masks 1,2,7,15) -----
#if __has_builtin(__builtin_amdgcn_update_dpp)
template <int CTRL>
__device__ __forceinline__ float dpp_movf(float x) {
  int i = __builtin_bit_cast(int, x);
  int r = __builtin_amdgcn_update_dpp(i, i, CTRL, 0xF, 0xF, true);
  return __builtin_bit_cast(float, r);
}
__device__ __forceinline__ float half_max16(float x) {
  x = fmaxf(x, dpp_movf<0xB1>(x));    // quad_perm(1,0,3,2) : xor 1
  x = fmaxf(x, dpp_movf<0x4E>(x));    // quad_perm(2,3,0,1) : xor 2
  x = fmaxf(x, dpp_movf<0x141>(x));   // row_half_mirror    : xor 7
  x = fmaxf(x, dpp_movf<0x140>(x));   // row_mirror         : xor 15
  return x;
}
__device__ __forceinline__ float half_sum16(float x) {
  x += dpp_movf<0xB1>(x);
  x += dpp_movf<0x4E>(x);
  x += dpp_movf<0x141>(x);
  x += dpp_movf<0x140>(x);
  return x;
}
#else
__device__ __forceinline__ float half_max16(float x) {
  x = fmaxf(x, __shfl_xor(x, 1, 32));
  x = fmaxf(x, __shfl_xor(x, 2, 32));
  x = fmaxf(x, __shfl_xor(x, 4, 32));
  x = fmaxf(x, __shfl_xor(x, 8, 32));
  return x;
}
__device__ __forceinline__ float half_sum16(float x) {
  x += __shfl_xor(x, 1, 32);
  x += __shfl_xor(x, 2, 32);
  x += __shfl_xor(x, 4, 32);
  x += __shfl_xor(x, 8, 32);
  return x;
}
#endif

// ------------------------------ fp32 -> bf16 -------------------------------
__global__ __launch_bounds__(256)
void cvt_f32_bf16_x4(const float* __restrict__ src, bf16_t* __restrict__ dst,
                     int n4) {
  int i = blockIdx.x * blockDim.x + threadIdx.x;
  if (i < n4) {
    v4f f = *reinterpret_cast<const v4f*>(src + (size_t)i * 4);
    v4bf o;
    o[0] = (__bf16)f[0]; o[1] = (__bf16)f[1];
    o[2] = (__bf16)f[2]; o[3] = (__bf16)f[3];
    *reinterpret_cast<v4bf*>(dst + (size_t)i * 4) = o;
  }
}

// ------------------------------ WMMA GEMM ----------------------------------
// Y[m,n] = sum_k A[m,k] * W[n,k] + bias[n]     (torch Linear: x @ W.T + b)
// Block = 8 waves: wave w owns m-tile (blk_m*8 + w), all share col-group ng.
// The 64x32 W tile per k-step is async-staged into double-buffered LDS.
// MODE 0: bf16 -> [B,H,S,Hd]   MODE 1: bf16 -> [B,H,Hd,S]   MODE 2: f32 [M,768]
#define WSTRIDE 40   // LDS row stride (elements): 80B, 16B aligned, conflict-free

template <int MODE>
__global__ __launch_bounds__(256)
void gemm_bf16_wmma(const bf16_t* __restrict__ A, const bf16_t* __restrict__ W,
                    const float* __restrict__ bias, bf16_t* __restrict__ outb,
                    float* __restrict__ outf) {
  __shared__ bf16_t sW[2][64 * WSTRIDE];

  const int tid  = threadIdx.x;
  const int wave = tid >> 5;
  const int lane = tid & 31;
  const int l16  = lane & 15;
  const int hi   = lane >> 4;

  const int ng  = blockIdx.x % 12;           // 12 col-groups of 64
  const int mtg = blockIdx.x / 12;           // 64 groups of 8 m-tiles
  const int m0  = (mtg * 8 + wave) * 16;
  const int n0  = ng * 64;

  // staging map: 64 rows x 4 segments of 16B, one b128 per thread per step
  const int wrow = tid >> 2;                 // 0..63 (row n0+wrow of W)
  const int wseg = tid & 3;                  // 16B segment in the 32-wide k slice
  const bf16_t* wg_base = W + (size_t)(n0 + wrow) * ND + wseg * 8;
  bf16_t* wl_base = (bf16_t*)&sW[0][0] + wrow * WSTRIDE + wseg * 8;
  const int wbufstride = 64 * WSTRIDE;

  float bs[4];
#pragma unroll
  for (int t = 0; t < 4; ++t) bs[t] = bias[n0 + t * 16 + l16];

  v8f zero = {0.f, 0.f, 0.f, 0.f, 0.f, 0.f, 0.f, 0.f};
  v8f acc[4] = {zero, zero, zero, zero};

  const bf16_t* Arow = A + (size_t)(m0 + l16) * ND;   // lane's A row (M = l16)

  // prologue: stage k-step 0 into buffer 0
  copy16_g2l(wl_base, wg_base);
  wait_async_lds();
  __syncthreads();

  for (int step = 0; step < 24; ++step) {     // 768 / 32
    const int k0  = step * 32;
    const int cur = step & 1;
    if (step + 1 < 24)
      copy16_g2l(wl_base + (cur ^ 1) * wbufstride, wg_base + (k0 + 32));

    // A fragment: e=0..7 -> K=k0+hi*8+e ; e=8..15 -> K=k0+16+hi*8+(e-8)
    v16bf a = pack16(*reinterpret_cast<const v8bf*>(Arow + k0 + hi * 8),
                     *reinterpret_cast<const v8bf*>(Arow + k0 + 16 + hi * 8));
    const bf16_t* Wb = &sW[cur][0];

    // batch-load all four B fragments, then run the WMMA group
    v16bf bf[4];
#pragma unroll
    for (int t = 0; t < 4; ++t)
      bf[t] = load_frag(Wb + (t * 16 + l16) * WSTRIDE + hi * 8);
#pragma unroll
    for (int t = 0; t < 4; ++t)
      acc[t] = wmma_bf16(a, bf[t], acc[t]);
    pipeline_ds_wmma();

    wait_async_lds();
    __syncthreads();
  }

  // C/D layout: VGPR r, lane -> row m0 + r + 8*hi, col n0 + t*16 + l16
#pragma unroll
  for (int r = 0; r < 8; ++r) {
    const int m  = m0 + r + hi * 8;
    const int bb = m >> 10;
    const int s  = m & 1023;
#pragma unroll
    for (int t = 0; t < 4; ++t) {
      const int n   = n0 + t * 16 + l16;
      const float v = acc[t][r] + bs[t];
      if (MODE == 0) {
        const int h = n >> 6, d = n & 63;
        outb[(((size_t)(bb * NH + h)) * NS + s) * NHD + d] = (__bf16)v;
      } else if (MODE == 1) {
        const int h = n >> 6, d = n & 63;
        outb[(((size_t)(bb * NH + h)) * NHD + d) * NS + s] = (__bf16)v;
      } else {
        outf[(size_t)m * ND + n] = v;
      }
    }
  }
}

// --------------------------- flash attention -------------------------------
// Block = one (b,h) x 8 q-tiles (one per wave). K/V 32-key blocks are async-
// staged into double-buffered LDS shared by all 8 waves; online softmax uses
// DPP butterflies; P goes through a per-wave LDS tile to become an A-fragment.
#define PSTRIDE 40   // 80B rows
#define KSTRIDE 72   // 144B rows (32 rows x 64 elems + pad)
#define VSTRIDE 40   // 80B rows  (64 rows x 32 elems + pad)

__global__ __launch_bounds__(256)
void attn_wmma(const bf16_t* __restrict__ Q, const bf16_t* __restrict__ K,
               const bf16_t* __restrict__ Vt, bf16_t* __restrict__ AO) {
  __shared__ bf16_t sK[2][32 * KSTRIDE];
  __shared__ bf16_t sV[2][64 * VSTRIDE];
  __shared__ bf16_t sP[8][16 * PSTRIDE];

  const int tid  = threadIdx.x;
  const int wave = tid >> 5;
  const int lane = tid & 31;
  const int l16  = lane & 15;
  const int hi   = lane >> 4;
  bf16_t* P = &sP[wave][0];

  const int bh = blockIdx.x >> 3;            // 0..95  (b*12+h)
  const int qg = blockIdx.x & 7;             // q-tile group
  const int b  = bh / NH;
  const int h  = bh % NH;
  const int q0 = (qg * 8 + wave) * 16;

  const bf16_t* Qh = Q  + (size_t)bh * NS * NHD;
  const bf16_t* Kh = K  + (size_t)bh * NS * NHD;
  const bf16_t* Vh = Vt + (size_t)bh * NHD * NS;

  // staging maps (one async b128 per thread per tensor per step)
  const int krow = tid >> 3, kseg = tid & 7;   // K: 32 rows x 8 x 16B
  const int vrow = tid >> 2, vseg = tid & 3;   // V: 64 rows x 4 x 16B
  const bf16_t* kg_base = Kh + (size_t)krow * NHD + kseg * 8;   // + kb*NHD
  const bf16_t* vg_base = Vh + (size_t)vrow * NS + vseg * 8;    // + kb
  bf16_t* kl = (bf16_t*)&sK[0][0] + krow * KSTRIDE + kseg * 8;
  bf16_t* vl = (bf16_t*)&sV[0][0] + vrow * VSTRIDE + vseg * 8;
  const int kbufstride = 32 * KSTRIDE;
  const int vbufstride = 64 * VSTRIDE;

  // Q A-fragments for the two 32-wide K-slices of Hd=64
  const bf16_t* Qrow = Qh + (size_t)(q0 + l16) * NHD;
  v16bf qa[2];
#pragma unroll
  for (int s2 = 0; s2 < 2; ++s2)
    qa[s2] = pack16(*reinterpret_cast<const v8bf*>(Qrow + s2 * 32 + hi * 8),
                    *reinterpret_cast<const v8bf*>(Qrow + s2 * 32 + 16 + hi * 8));

  v8f zero = {0.f, 0.f, 0.f, 0.f, 0.f, 0.f, 0.f, 0.f};
  v8f o[4] = {zero, zero, zero, zero};        // 16x64 f32 output accumulator
  float mrow[8], lrow[8];
#pragma unroll
  for (int r = 0; r < 8; ++r) { mrow[r] = -1e30f; lrow[r] = 0.f; }
  const float scale = 0.125f;                 // Hd^-0.5

  // prologue: stage key-block 0 into buffer 0
  copy16_g2l(kl, kg_base);
  copy16_g2l(vl, vg_base);
  wait_async_lds();
  __syncthreads();

  for (int it = 0; it < 32; ++it) {           // 1024 / 32 key blocks
    const int kb  = it * 32;
    const int cur = it & 1;
    if (it + 1 < 32) {
      copy16_g2l(kl + (cur ^ 1) * kbufstride, kg_base + (size_t)(kb + 32) * NHD);
      copy16_g2l(vl + (cur ^ 1) * vbufstride, vg_base + (kb + 32));
    }
    const bf16_t* Kb = &sK[cur][0];
    const bf16_t* Vb = &sV[cur][0];

    // ---- scores: two 16x16 tiles (keys kb..kb+15, kb+16..kb+31) ----
    const bf16_t* K0 = Kb + l16 * KSTRIDE;          // key kb + l16
    const bf16_t* K1 = Kb + (16 + l16) * KSTRIDE;   // key kb + 16 + l16

    // batch-load all four K B-fragments, then run the WMMA group
    v16bf kb0[2], kb1[2];
#pragma unroll
    for (int s2 = 0; s2 < 2; ++s2) {
      kb0[s2] = load_frag(K0 + s2 * 32 + hi * 8);
      kb1[s2] = load_frag(K1 + s2 * 32 + hi * 8);
    }
    v8f s0 = zero, s1 = zero;
#pragma unroll
    for (int s2 = 0; s2 < 2; ++s2) {
      s0 = wmma_bf16(qa[s2], kb0[s2], s0);
      s1 = wmma_bf16(qa[s2], kb1[s2], s1);
    }
    pipeline_ds_wmma();

    // ---- online softmax per row (row's 16 cols live in one half-wave) ----
#pragma unroll
    for (int r = 0; r < 8; ++r) {
      float a0 = s0[r] * scale, a1 = s1[r] * scale;
      float bm = half_max16(fmaxf(a0, a1));
      const float mnew  = fmaxf(mrow[r], bm);
      const float alpha = __expf(mrow[r] - mnew);
      const float p0 = __expf(a0 - mnew);
      const float p1 = __expf(a1 - mnew);
      const float rs = half_sum16(p0 + p1);
      lrow[r] = lrow[r] * alpha + rs;
      mrow[r] = mnew;
      o[0][r] *= alpha; o[1][r] *= alpha; o[2][r] *= alpha; o[3][r] *= alpha;
      const int row = r + hi * 8;             // C-layout row for this VGPR/half
      P[row * PSTRIDE + l16]      = (__bf16)p0;
      P[row * PSTRIDE + 16 + l16] = (__bf16)p1;
    }

    // same-wave LDS RAW: DS is in-order per wave; fence compiler + drain DScnt
    asm volatile("s_wait_dscnt 0" ::: "memory");

    // ---- O += P(16x32) * V(32x64) ----
    v16bf pa = pack16(*reinterpret_cast<const v8bf*>(P + l16 * PSTRIDE + hi * 8),
                      *reinterpret_cast<const v8bf*>(P + l16 * PSTRIDE + 16 + hi * 8));

    // batch-load all four V B-fragments, then run the WMMA group
    v16bf vbf[4];
#pragma unroll
    for (int t = 0; t < 4; ++t)
      vbf[t] = load_frag(Vb + (t * 16 + l16) * VSTRIDE + hi * 8);
#pragma unroll
    for (int t = 0; t < 4; ++t)
      o[t] = wmma_bf16(pa, vbf[t], o[t]);
    pipeline_ds_wmma();

    wait_async_lds();       // next buffer resident before anyone flips to it
    __syncthreads();        // and everyone done reading this buffer
  }

  // ---- normalize and store to [B,S,D] bf16 ----
#pragma unroll
  for (int r = 0; r < 8; ++r) {
    const float inv = 1.0f / lrow[r];
    const int s = q0 + r + hi * 8;
    bf16_t* orow = AO + ((size_t)(b * NS + s)) * ND + h * NHD;
#pragma unroll
    for (int t = 0; t < 4; ++t)
      orow[t * 16 + l16] = (__bf16)(o[t][r] * inv);
  }
}

// ------------------------------- launcher ----------------------------------
extern "C" void kernel_launch(void* const* d_in, const int* in_sizes, int n_in,
                              void* d_out, int out_size, void* d_ws,
                              size_t ws_size, hipStream_t stream) {
  const float* x  = (const float*)d_in[0];
  const float* Wq = (const float*)d_in[1];
  const float* bq = (const float*)d_in[2];
  const float* Wk = (const float*)d_in[3];
  const float* bk = (const float*)d_in[4];
  const float* Wv = (const float*)d_in[5];
  const float* bv = (const float*)d_in[6];
  const float* Wp = (const float*)d_in[7];
  const float* bp = (const float*)d_in[8];
  float* out = (float*)d_out;

  char* ws = (char*)d_ws;
  size_t off = 0;
  auto carve = [&](size_t bytes) -> char* {
    char* p = ws + off;
    off += (bytes + 255) & ~(size_t)255;
    return p;
  };
  bf16_t* xbf  = (bf16_t*)carve((size_t)NM * ND * 2);
  bf16_t* wqb  = (bf16_t*)carve((size_t)ND * ND * 2);
  bf16_t* wkb  = (bf16_t*)carve((size_t)ND * ND * 2);
  bf16_t* wvb  = (bf16_t*)carve((size_t)ND * ND * 2);
  bf16_t* wpb  = (bf16_t*)carve((size_t)ND * ND * 2);
  bf16_t* Qbf  = (bf16_t*)carve((size_t)NM * ND * 2);   // [B,H,S,Hd]
  bf16_t* Kbf  = (bf16_t*)carve((size_t)NM * ND * 2);   // [B,H,S,Hd]
  bf16_t* Vtbf = (bf16_t*)carve((size_t)NM * ND * 2);   // [B,H,Hd,S]
  bf16_t* AObf = (bf16_t*)carve((size_t)NM * ND * 2);   // [B,S,D]

  // fp32 -> bf16 staging
  {
    const int n4x = NM * ND / 4;
    cvt_f32_bf16_x4<<<(n4x + 255) / 256, 256, 0, stream>>>(x, xbf, n4x);
    const int n4w = ND * ND / 4;
    const int gw  = (n4w + 255) / 256;
    cvt_f32_bf16_x4<<<gw, 256, 0, stream>>>(Wq, wqb, n4w);
    cvt_f32_bf16_x4<<<gw, 256, 0, stream>>>(Wk, wkb, n4w);
    cvt_f32_bf16_x4<<<gw, 256, 0, stream>>>(Wv, wvb, n4w);
    cvt_f32_bf16_x4<<<gw, 256, 0, stream>>>(Wp, wpb, n4w);
  }

  // QKV projections: 768 blocks = 64 m-groups x 12 col-groups
  gemm_bf16_wmma<0><<<768, 256, 0, stream>>>(xbf, wqb, bq, Qbf, nullptr);
  gemm_bf16_wmma<0><<<768, 256, 0, stream>>>(xbf, wkb, bk, Kbf, nullptr);
  gemm_bf16_wmma<1><<<768, 256, 0, stream>>>(xbf, wvb, bv, Vtbf, nullptr);

  // attention: 768 blocks = 96 (b,h) x 8 q-tile groups
  attn_wmma<<<768, 256, 0, stream>>>(Qbf, Kbf, Vtbf, AObf);

  // output projection, fp32 straight to d_out
  gemm_bf16_wmma<2><<<768, 256, 0, stream>>>(AObf, wpb, bp, nullptr, out);
}
